// SentenceGraphGNN_91311004713454
// MI455X (gfx1250) — matrix operand order repaired and checked by reading, not compiled, not run on hardware
//
#include <hip/hip_runtime.h>
#include <hip/hip_bf16.h>
#include <math.h>

#define NNODE 50000
#define NEDGE 500000
#define INF_D 384
#define HIDD  128
#define NHEAD 8
#define CHAN  16
#define NLAY  3
#define OUTC  16
#define NEG_SLOPE 0.2f

typedef __attribute__((ext_vector_type(16))) __bf16 v16bf;
typedef __attribute__((ext_vector_type(8)))  __bf16 v8bf;
typedef __attribute__((ext_vector_type(8)))  float  v8f;

__device__ __forceinline__ __bf16 f2bf(float f) {
    union { float f; unsigned u; } a; a.f = f;
    unsigned u = a.u;
    unsigned r = u + 0x7FFFu + ((u >> 16) & 1u);   // round-to-nearest-even
    union { unsigned short s; __bf16 b; } o; o.s = (unsigned short)(r >> 16);
    return o.b;
}
__device__ __forceinline__ v16bf cat8(v8bf lo, v8bf hi) {
    return __builtin_shufflevector(lo, hi, 0,1,2,3,4,5,6,7,8,9,10,11,12,13,14,15);
}

// monotonic float<->uint mapping for atomicMax-based segment max
__device__ __forceinline__ unsigned ford(float f) {
    union { float f; unsigned u; } a; a.f = f;
    return (a.u >> 31) ? ~a.u : (a.u | 0x80000000u);
}
__device__ __forceinline__ float fdec(unsigned o) {
    union { unsigned u; float f; } a;
    a.u = (o >> 31) ? (o ^ 0x80000000u) : ~o;
    return a.f;
}

// ---------------------------------------------------------------------------
// WMMA GEMM: C[M, NT*16] = act(A[M,K] @ Bt[NT*16,K]^T + bias).
// A, Bt are bf16; one wave per 16-row strip, covering all NT n-tiles so the
// A fragment is loaded once per k-step. All fragment loads are b128.
// Fragment layouts per ISA 7.12.2:
//   A 16x32:  lane m=lane&15, half=lane>>4; VGPR v -> K = {h*8+2v%8.., 16+h*8+..}
//             == two contiguous 8-elem runs at k0+h*8 and k0+16+h*8.
//   B 32x16:  lane n=lane&15; VGPR v -> K = {h*8+2v%8.., 16+h*8+..}
//             == two contiguous 8-elem runs at k0+h*8 and k0+16+h*8 (row n of Bt).
// ---------------------------------------------------------------------------
template<int NT>
__global__ void wmma_gemm_bf16(const __bf16* __restrict__ A,
                               const __bf16* __restrict__ Bt,
                               const float* __restrict__ bias,
                               float* __restrict__ C,
                               int M, int K, int relu) {
    constexpr int Nn = NT * 16;
    int wave = (blockIdx.x * blockDim.x + threadIdx.x) >> 5;  // = m-tile
    int lane = threadIdx.x & 31;
    if (wave * 16 >= M) return;
    int half = lane >> 4;
    int l15  = lane & 15;
    const __bf16* Arow = A + (size_t)(wave * 16 + l15) * K;

    v8f acc[NT];
#pragma unroll
    for (int t = 0; t < NT; ++t) acc[t] = (v8f){};

    for (int k0 = 0; k0 < K; k0 += 32) {
        v8bf alo = *(const v8bf*)(Arow + k0 + half * 8);
        v8bf ahi = *(const v8bf*)(Arow + k0 + 16 + half * 8);
        v16bf afrag = cat8(alo, ahi);
#pragma unroll
        for (int t = 0; t < NT; ++t) {
            const __bf16* Brow = Bt + (size_t)(t * 16 + l15) * K + k0 + half * 8;
            v16bf bfrag = cat8(*(const v8bf*)Brow, *(const v8bf*)(Brow + 16));
            acc[t] = __builtin_amdgcn_wmma_f32_16x16x32_bf16(
                false, afrag, false, bfrag, (short)0, acc[t], false, false);
        }
    }
#pragma unroll
    for (int t = 0; t < NT; ++t) {
        int ncol = t * 16 + l15;
        float bv = bias ? bias[ncol] : 0.f;
#pragma unroll
        for (int r = 0; r < 8; ++r) {              // D row = r + 8*half
            int m = wave * 16 + r + (half << 3);
            float v = acc[t][r] + bv;
            if (relu) v = fmaxf(v, 0.f);
            C[(size_t)m * Nn + ncol] = v;
        }
    }
}

// ---------------------------------------------------------------------------
__global__ void fill_u32(unsigned* __restrict__ p, unsigned v, int n) {
    int i = blockIdx.x * blockDim.x + threadIdx.x;
    if (i < n) p[i] = v;
}

__global__ void convert_f32_bf16(const float* __restrict__ src,
                                 __bf16* __restrict__ dst, int n) {
    int i = blockIdx.x * blockDim.x + threadIdx.x;
    if (i < n) dst[i] = f2bf(src[i]);
}

// Bt[n*K+k] = bf16(B[k*Nn+n])
__global__ void transpose_f32_bf16(const float* __restrict__ B,
                                   __bf16* __restrict__ Bt, int K, int Nn) {
    int i = blockIdx.x * blockDim.x + threadIdx.x;
    if (i >= K * Nn) return;
    int k = i / Nn, n = i - k * Nn;
    Bt[(size_t)n * K + k] = f2bf(B[i]);
}

// al_s / al_d : [N,H] = einsum('nhc,hc->nh')
__global__ void att_logits_k(const float* __restrict__ xh,
                             const float* __restrict__ a_src,
                             const float* __restrict__ a_dst,
                             float* __restrict__ al_s,
                             float* __restrict__ al_d) {
    int i = blockIdx.x * blockDim.x + threadIdx.x;   // n*H + h
    if (i >= NNODE * NHEAD) return;
    int h = i & (NHEAD - 1);
    const float* xp = xh + (size_t)i * CHAN;
    float ss = 0.f, sd = 0.f;
#pragma unroll
    for (int c = 0; c < CHAN; ++c) {
        float v = xp[c];
        ss += v * a_src[h * CHAN + c];
        sd += v * a_dst[h * CHAN + c];
    }
    al_s[i] = ss; al_d[i] = sd;
}

// leaky_relu(al_s[src]+al_d[dst]) per (edge,head); segment max into mord
__global__ void edge_logits_max_k(const int* __restrict__ ei,
                                  const float* __restrict__ al_s,
                                  const float* __restrict__ al_d,
                                  float* __restrict__ e_out,
                                  unsigned* __restrict__ mord) {
    int idx = blockIdx.x * blockDim.x + threadIdx.x;
    if (idx >= (NEDGE + NNODE) * NHEAD) return;
    int eidx = idx >> 3, h = idx & 7;
    int s, d;
    if (eidx < NEDGE) {
        s = ei[eidx]; d = ei[NEDGE + eidx];
        __builtin_prefetch(&ei[eidx + 2048], 0, 0);          // stream edge indices
    } else {
        s = d = eidx - NEDGE;                                 // self loops
    }
    float v = al_s[s * NHEAD + h] + al_d[d * NHEAD + h];
    v = (v > 0.f) ? v : NEG_SLOPE * v;
    e_out[idx] = v;
    atomicMax(&mord[d * NHEAD + h], ford(v));
}

// e <- exp(e - m[dst]); denom[dst] += e
__global__ void edge_exp_sum_k(const int* __restrict__ ei,
                               const unsigned* __restrict__ mord,
                               float* __restrict__ e_out,
                               float* __restrict__ denom) {
    int idx = blockIdx.x * blockDim.x + threadIdx.x;
    if (idx >= (NEDGE + NNODE) * NHEAD) return;
    int eidx = idx >> 3, h = idx & 7;
    int d = (eidx < NEDGE) ? ei[NEDGE + eidx] : (eidx - NEDGE);
    float m  = fdec(mord[d * NHEAD + h]);
    float ex = __expf(e_out[idx] - m);
    e_out[idx] = ex;
    atomicAdd(&denom[d * NHEAD + h], ex);
}

// acc[dst, j] += xh[src, j] * alpha(edge, head(j))
__global__ void edge_scatter_k(const int* __restrict__ ei,
                               const float* __restrict__ e_out,
                               const float* __restrict__ denom,
                               const float* __restrict__ xh,
                               float* __restrict__ acc) {
    int idx = blockIdx.x * blockDim.x + threadIdx.x;
    if (idx >= (NEDGE + NNODE) * HIDD) return;
    int eidx = idx >> 7, j = idx & 127, h = j >> 4;
    int s, d;
    if (eidx < NEDGE) { s = ei[eidx]; d = ei[NEDGE + eidx]; }
    else              { s = d = eidx - NEDGE; }
    float alpha = e_out[eidx * NHEAD + h] / (denom[d * NHEAD + h] + 1e-16f);
    atomicAdd(&acc[(size_t)d * HIDD + j], xh[(size_t)s * HIDD + j] * alpha);
}

// h_out = relu(layernorm(acc + gat_b + h_res) * g + b) ; one wave32 per node
__global__ void resid_ln_relu_k(const float* __restrict__ acc,
                                const float* __restrict__ hres,
                                const float* __restrict__ gat_b,
                                const float* __restrict__ g,
                                const float* __restrict__ b,
                                float* __restrict__ hout) {
    int wave = (blockIdx.x * blockDim.x + threadIdx.x) >> 5;
    int lane = threadIdx.x & 31;
    if (wave >= NNODE) return;
    size_t rowo = (size_t)wave * HIDD + lane * 4;
    float4 av = *(const float4*)(acc + rowo);
    float4 hv = *(const float4*)(hres + rowo);
    float4 bb = *(const float4*)(gat_b + lane * 4);
    float v[4] = { av.x + bb.x + hv.x, av.y + bb.y + hv.y,
                   av.z + bb.z + hv.z, av.w + bb.w + hv.w };
    float sum = v[0] + v[1] + v[2] + v[3];
    for (int off = 16; off > 0; off >>= 1) sum += __shfl_xor(sum, off, 32);
    float mu = sum * (1.f / HIDD);
    float var = 0.f;
#pragma unroll
    for (int i = 0; i < 4; ++i) { float dlt = v[i] - mu; var += dlt * dlt; }
    for (int off = 16; off > 0; off >>= 1) var += __shfl_xor(var, off, 32);
    float rinv = rsqrtf(var * (1.f / HIDD) + 1e-5f);
    float4 gv = *(const float4*)(g + lane * 4);
    float4 be = *(const float4*)(b + lane * 4);
    float4 o;
    o.x = fmaxf((v[0] - mu) * rinv * gv.x + be.x, 0.f);
    o.y = fmaxf((v[1] - mu) * rinv * gv.y + be.y, 0.f);
    o.z = fmaxf((v[2] - mu) * rinv * gv.z + be.z, 0.f);
    o.w = fmaxf((v[3] - mu) * rinv * gv.w + be.w, 0.f);
    *(float4*)(hout + rowo) = o;
}

__global__ void logsoftmax16_k(const float* __restrict__ logits,
                               float* __restrict__ out) {
    int n = blockIdx.x * blockDim.x + threadIdx.x;
    if (n >= NNODE) return;
    const float* p = logits + (size_t)n * OUTC;
    float mx = p[0];
#pragma unroll
    for (int k = 1; k < OUTC; ++k) mx = fmaxf(mx, p[k]);
    float s = 0.f;
#pragma unroll
    for (int k = 0; k < OUTC; ++k) s += __expf(p[k] - mx);
    float lse = mx + __logf(s);
#pragma unroll
    for (int k = 0; k < OUTC; ++k) out[(size_t)n * OUTC + k] = p[k] - lse;
}

// ---------------------------------------------------------------------------
extern "C" void kernel_launch(void* const* d_in, const int* in_sizes, int n_in,
                              void* d_out, int out_size, void* d_ws, size_t ws_size,
                              hipStream_t stream) {
    const float* x       = (const float*)d_in[0];
    const int*   ei      = (const int*)  d_in[1];
    const float* proj_w  = (const float*)d_in[2];
    const float* proj_b  = (const float*)d_in[3];
    const float* lin_w   = (const float*)d_in[4];
    const float* att_src = (const float*)d_in[5];
    const float* att_dst = (const float*)d_in[6];
    const float* gat_b   = (const float*)d_in[7];
    const float* ln_g    = (const float*)d_in[8];
    const float* ln_b    = (const float*)d_in[9];
    const float* cls_w1  = (const float*)d_in[10];
    const float* cls_b1  = (const float*)d_in[11];
    const float* cls_w2  = (const float*)d_in[12];
    const float* cls_b2  = (const float*)d_in[13];
    float* out = (float*)d_out;

    // ---- workspace layout -------------------------------------------------
    char* base = (char*)d_ws;
    size_t off = 0;
    auto alloc = [&](size_t bytes) -> void* {
        void* r = base + off;
        off += bytes;
        off = (off + 255) & ~(size_t)255;
        return r;
    };
    float*    hA    = (float*)   alloc((size_t)NNODE * HIDD * 4);
    float*    hB    = (float*)   alloc((size_t)NNODE * HIDD * 4);
    float*    xh    = (float*)   alloc((size_t)NNODE * HIDD * 4);   // adjacent to acc
    float*    acc   = (float*)   alloc((size_t)NNODE * HIDD * 4);
    float*    e_buf = (float*)   alloc((size_t)(NEDGE + NNODE) * NHEAD * 4);
    unsigned* mord  = (unsigned*)alloc((size_t)NNODE * NHEAD * 4);
    float*    denom = (float*)   alloc((size_t)NNODE * NHEAD * 4);
    float*    al_s  = (float*)   alloc((size_t)NNODE * NHEAD * 4);
    float*    al_d  = (float*)   alloc((size_t)NNODE * NHEAD * 4);
    __bf16*   h_bf  = (__bf16*)  alloc((size_t)NNODE * HIDD * 2);
    __bf16*   pwt   = (__bf16*)  alloc((size_t)INF_D * HIDD * 2);        // proj_w^T
    __bf16*   lwt   = (__bf16*)  alloc((size_t)NLAY * HIDD * HIDD * 2);  // lin_w^T
    __bf16*   w1t   = (__bf16*)  alloc((size_t)HIDD * (HIDD / 2) * 2);
    __bf16*   w2t   = (__bf16*)  alloc((size_t)(HIDD / 2) * OUTC * 2);
    // aliases over dead regions:
    __bf16* x_bf   = (__bf16*)xh;     // 38.4MB <= xh+acc (51.2MB); dead before use
    float*  mid    = xh;              // classifier hidden (dead after layers)
    __bf16* mid_bf = (__bf16*)e_buf;  // 6.4MB  <= e_buf (17.6MB)
    float*  logits = acc;

    const int TB = 256;
    auto cdiv = [](long long a, long long b) { return (int)((a + b - 1) / b); };
    auto gemm = [&](int NT, const __bf16* A, const __bf16* Bt, const float* bias,
                    float* C, int M, int K, int relu) {
        int blocks = cdiv((long long)(M / 16) * 32, TB);
        if (NT == 8)      wmma_gemm_bf16<8><<<blocks, TB, 0, stream>>>(A, Bt, bias, C, M, K, relu);
        else if (NT == 4) wmma_gemm_bf16<4><<<blocks, TB, 0, stream>>>(A, Bt, bias, C, M, K, relu);
        else              wmma_gemm_bf16<1><<<blocks, TB, 0, stream>>>(A, Bt, bias, C, M, K, relu);
    };
    auto fill = [&](void* p, unsigned v, size_t cnt) {
        fill_u32<<<cdiv(cnt, TB), TB, 0, stream>>>((unsigned*)p, v, (int)cnt);
    };
    auto tobf = [&](const float* s, __bf16* d, size_t cnt) {
        convert_f32_bf16<<<cdiv(cnt, TB), TB, 0, stream>>>(s, d, (int)cnt);
    };
    auto transp = [&](const float* B, __bf16* Bt, int K, int Nn) {
        transpose_f32_bf16<<<cdiv((long long)K * Nn, TB), TB, 0, stream>>>(B, Bt, K, Nn);
    };

    // ---- weight prep (bf16, transposed) ------------------------------------
    transp(proj_w, pwt, INF_D, HIDD);
    for (int l = 0; l < NLAY; ++l)
        transp(lin_w + (size_t)l * HIDD * HIDD, lwt + (size_t)l * HIDD * HIDD, HIDD, HIDD);
    transp(cls_w1, w1t, HIDD, HIDD / 2);
    transp(cls_w2, w2t, HIDD / 2, OUTC);

    // ---- h = relu(x @ proj_w + proj_b) -------------------------------------
    tobf(x, x_bf, (size_t)NNODE * INF_D);
    gemm(8, x_bf, pwt, proj_b, hA, NNODE, INF_D, 1);

    float* h_cur = hA;
    float* h_nxt = hB;
    const int EH = (NEDGE + NNODE) * NHEAD;
    const int EC = (NEDGE + NNODE) * HIDD;

    for (int l = 0; l < NLAY; ++l) {
        fill(acc,   0u, (size_t)NNODE * HIDD);
        fill(mord,  0u, (size_t)NNODE * NHEAD);   // ordered-min (< any real value)
        fill(denom, 0u, (size_t)NNODE * NHEAD);

        tobf(h_cur, h_bf, (size_t)NNODE * HIDD);
        gemm(8, h_bf, lwt + (size_t)l * HIDD * HIDD, nullptr, xh, NNODE, HIDD, 0);

        att_logits_k<<<cdiv(NNODE * NHEAD, TB), TB, 0, stream>>>(
            xh, att_src + (size_t)l * NHEAD * CHAN, att_dst + (size_t)l * NHEAD * CHAN,
            al_s, al_d);

        edge_logits_max_k<<<cdiv(EH, TB), TB, 0, stream>>>(ei, al_s, al_d, e_buf, mord);
        edge_exp_sum_k   <<<cdiv(EH, TB), TB, 0, stream>>>(ei, mord, e_buf, denom);
        edge_scatter_k   <<<cdiv(EC, TB), TB, 0, stream>>>(ei, e_buf, denom, xh, acc);

        resid_ln_relu_k<<<cdiv(NNODE * 32, TB), TB, 0, stream>>>(
            acc, h_cur, gat_b + (size_t)l * HIDD,
            ln_g + (size_t)l * HIDD, ln_b + (size_t)l * HIDD, h_nxt);

        float* t = h_cur; h_cur = h_nxt; h_nxt = t;
    }

    // ---- classifier --------------------------------------------------------
    tobf(h_cur, h_bf, (size_t)NNODE * HIDD);
    gemm(4, h_bf, w1t, cls_b1, mid, NNODE, HIDD, 1);
    tobf(mid, mid_bf, (size_t)NNODE * (HIDD / 2));
    gemm(1, mid_bf, w2t, cls_b2, logits, NNODE, HIDD / 2, 0);
    logsoftmax16_k<<<cdiv(NNODE, TB), TB, 0, stream>>>(logits, out);
}
